// FcGagaLayer_87574383165571
// MI455X (gfx1250) — compile-verified, matrix operand order checked by reading, not run on hardware
//
#include <hip/hip_runtime.h>

// Problem constants (FC-GAGA reference)
#define BB    32
#define NNODE 325
#define HH    12
#define HORZ  12
#define DD    64
#define HUU   128
#define NBB   3
#define ININ  3976          // HH + NNODE*HH + DD
#define KACAT 96            // DD+HH=76 zero-padded so bf16 rows are 16B-aligned
#define EPSV  10.0f

typedef __attribute__((ext_vector_type(16))) __bf16 v16bf;
typedef __attribute__((ext_vector_type(8)))  float  v8f;

// float bits + 0x8000: bf16 round-to-nearest (ties away); bf16 = bits[31:16].
__device__ __forceinline__ unsigned rnd16(float x) {
    return __builtin_bit_cast(unsigned, x) + 0x8000u;
}
__device__ __forceinline__ unsigned short bf16of(float x) {
    return (unsigned short)(rnd16(x) >> 16);
}

struct BfPair { uint4 lo, hi; };
__device__ __forceinline__ v16bf mkfrag(uint4 lo, uint4 hi) {
    BfPair p{lo, hi};
    return __builtin_bit_cast(v16bf, p);
}

// ---------------------------------------------------------------------------
// bf16-WMMA GEMM on pre-converted operands:
//   Abf: M x K bf16 row-major (activations)
//   BT : N x K bf16 row-major (weights pre-transposed+converted)
// C = f(A @ B + bias). 128x128 tile / 256 threads (8 wave32); K chunked by 32;
// software pipelined (stage next chunk into regs during the WMMA burst).
// Both LDS tiles are straight row copies (no transpose, no conversion):
// per chunk per thread: 2x global b128 + 2x ds_store b128 per operand.
// LDS row stride 20 dwords (80B: 16B-aligned, conflict-free walk mod 64 banks).
// mode 0: Cb = bf16(relu(v))            (activations, bf16-only output)
// mode 1: Cf += v                       (forecast accumulate)
// mode 2: bc=relu(Cf-v); Cf=bc; Cb=bf16(bc)   (backcast: fp32 + bf16 mirror)
// mode 3: Cf = v                        (plain fp32 store)
// ---------------------------------------------------------------------------
__global__ __launch_bounds__(256) void gemm_bf16_wmma(
    const unsigned short* __restrict__ Abf, const unsigned short* __restrict__ BT,
    const float* __restrict__ bias,
    float* __restrict__ Cf, unsigned short* __restrict__ Cb,
    int M, int N, int K, int mode)
{
    __shared__ unsigned Asu[128][20];   // [row][k-pair]
    __shared__ unsigned Bsu[128][20];   // [col][k-pair]

    const int tid  = threadIdx.x;
    const int lane = tid & 31;
    const int wv   = tid >> 5;          // wave id -> 16-col N strip
    const int half = lane >> 4;         // WMMA lane-half (K split)
    const int r16  = lane & 15;

    const int m0 = blockIdx.x * 128;
    const int n0 = blockIdx.y * 128;

    v8f acc[8];
#pragma unroll
    for (int t = 0; t < 8; ++t)
#pragma unroll
        for (int vr = 0; vr < 8; ++vr) acc[t][vr] = 0.f;

    uint4 ua[2], ub[2];                 // staged tile registers (8 bf16 each)

    // Stage a 128-row x 32-k bf16 panel (row-major, row stride K) into regs.
    auto stage = [&](const unsigned short* __restrict__ P, int rows0, int nrows,
                     int k0, uint4* rr) {
        if ((rows0 + 128) <= nrows && (k0 + 32) <= K) {
#pragma unroll
            for (int it = 0; it < 2; ++it) {
                int idx = tid + it * 256;          // 512 uint4 = 128 x 4
                int r = idx >> 2, q = idx & 3;
                rr[it] = *(const uint4*)(P + (size_t)(rows0 + r) * K + k0 + 8 * q);
            }
        } else {
#pragma unroll
            for (int it = 0; it < 2; ++it) {
                int idx = tid + it * 256;
                int r = idx >> 2, q = idx & 3;
                int gr = rows0 + r;
                unsigned e[8];
#pragma unroll
                for (int j = 0; j < 8; ++j) {
                    int gk = k0 + 8 * q + j;
                    bool ok = (gr < nrows) && (gk < K);
                    unsigned v = P[(size_t)(ok ? gr : 0) * K + (ok ? gk : 0)];
                    e[j] = ok ? v : 0u;
                }
                uint4 u;
                u.x = e[0] | (e[1] << 16); u.y = e[2] | (e[3] << 16);
                u.z = e[4] | (e[5] << 16); u.w = e[6] | (e[7] << 16);
                rr[it] = u;
            }
        }
    };
    auto commit = [&](unsigned (*S)[20], const uint4* rr) {
#pragma unroll
        for (int it = 0; it < 2; ++it) {
            int idx = tid + it * 256;
            int r = idx >> 2, q = idx & 3;
            *(uint4*)&S[r][q * 4] = rr[it];        // ds_store_b128, 16B-aligned
        }
    };

    const int nk = (K + 31) >> 5;
    stage(Abf, m0, M, 0, ua);
    stage(BT,  n0, N, 0, ub);
    for (int kt = 0; kt < nk; ++kt) {
        commit(Asu, ua);
        commit(Bsu, ub);
        __syncthreads();
        if (kt + 1 < nk) {
            int k1 = (kt + 1) << 5;
            stage(Abf, m0, M, k1, ua);             // prefetch under WMMA burst
            stage(BT,  n0, N, k1, ub);
        }
        // Fragments: WMMA 16-bit layout -> lane-half selects dword groups
        // lo = pairs k/2 in [4h,4h+4), hi = pairs in [8+4h, 8+4h+4).
        const int rowB = wv * 16 + r16;
        uint4 blo = *(const uint4*)&Bsu[rowB][4 * half];
        uint4 bhi = *(const uint4*)&Bsu[rowB][8 + 4 * half];
        v16bf vb = mkfrag(blo, bhi);
#pragma unroll
        for (int t = 0; t < 8; ++t) {
            const int rowA = t * 16 + r16;
            uint4 alo = *(const uint4*)&Asu[rowA][4 * half];
            uint4 ahi = *(const uint4*)&Asu[rowA][8 + 4 * half];
            v16bf va = mkfrag(alo, ahi);
            acc[t] = __builtin_amdgcn_wmma_f32_16x16x32_bf16(
                false, va, false, vb, (short)0, acc[t], false, false);
        }
        __syncthreads();
    }

    // Epilogue. D layout: VGPR vr, lane l -> m = vr + 8*(l>>4), n = l&15.
    const int gn = n0 + wv * 16 + r16;
    if (gn < N) {
        const float bv = bias ? bias[gn] : 0.f;
#pragma unroll
        for (int t = 0; t < 8; ++t) {
#pragma unroll
            for (int vr = 0; vr < 8; ++vr) {
                int gr = m0 + t * 16 + vr + 8 * half;
                if (gr < M) {
                    size_t o = (size_t)gr * N + gn;
                    float v = acc[t][vr] + bv;
                    if (mode == 0) {
                        float r = v > 0.f ? v : 0.f;
                        Cb[o] = bf16of(r);
                    } else if (mode == 3) {
                        Cf[o] = v;
                    } else if (mode == 1) {
                        Cf[o] += v;
                    } else {
                        float bc = Cf[o] - v;
                        bc = bc > 0.f ? bc : 0.f;
                        Cf[o] = bc;
                        Cb[o] = bf16of(bc);
                    }
                }
            }
        }
    }
}

// Weight transpose+convert: T[n*Kp + k] = bf16(W[k*N + n]) (zero-pad k >= K)
__global__ void wtrans_kernel(const float* __restrict__ W,
                              unsigned short* __restrict__ T,
                              int K, int N, int Kp) {
    int i = blockIdx.x * blockDim.x + threadIdx.x;
    if (i >= N * Kp) return;
    int n = i / Kp, k = i - n * Kp;
    float v = (k < K) ? W[(size_t)k * N + n] : 0.f;
    T[(size_t)n * Kp + k] = bf16of(v);
}

// dp[i][j] = exp(EPS * dot(emb[i], emb[j]))
__global__ void dp_kernel(const float* __restrict__ emb, float* __restrict__ dp) {
    int i = blockIdx.x;
    int j = blockIdx.y * blockDim.x + threadIdx.x;
    if (j >= NNODE) return;
    float s = 0.f;
#pragma unroll 8
    for (int d = 0; d < DD; ++d) s += emb[i * DD + d] * emb[j * DD + d];
    dp[i * NNODE + j] = __expf(EPSV * s);
}

// acat[bn] = bf16(concat(emb[node_id[bn]], tod[bn]))  -> (B*N, 96), zero-padded
__global__ void build_acat_kernel(const float* __restrict__ tod,
                                  const int* __restrict__ nid,
                                  const float* __restrict__ emb,
                                  unsigned short* __restrict__ acat) {
    int bn = blockIdx.x;
    int id = nid[bn];
    for (int c = threadIdx.x; c < KACAT; c += blockDim.x) {
        float v = (c < DD) ? emb[id * DD + c]
                : (c < DD + HH) ? tod[(size_t)bn * HH + (c - DD)] : 0.f;
        acat[(size_t)bn * KACAT + c] = bf16of(v);
    }
}

// hist = history/(1+tg_bwd); level = max_t hist; histN = div_no_nan(hist, level)
__global__ void gate_hist_kernel(const float* __restrict__ history_in,
                                 const float* __restrict__ tgb,
                                 float* __restrict__ hist, float* __restrict__ level,
                                 float* __restrict__ histN) {
    int bn = blockIdx.x * blockDim.x + threadIdx.x;
    if (bn >= BB * NNODE) return;
    float h[HH]; float L = -3.4e38f;
#pragma unroll
    for (int t = 0; t < HH; ++t) {
        float v = history_in[bn * HH + t] / (1.f + tgb[bn * HH + t]);
        h[t] = v; L = v > L ? v : L;
    }
    level[bn] = L;
    float inv = (L == 0.f) ? 0.f : 1.f / L;
#pragma unroll
    for (int t = 0; t < HH; ++t) {
        hist[bn * HH + t]  = h[t];
        histN[bn * HH + t] = (L == 0.f) ? 0.f : h[t] * inv;
    }
}

// Materialize x = [histN | relu(div_no_nan(hist_j*dp[n,j]-L, L)) | emb[id]]
// fp32 into d_out's backcast region + bf16 mirror for the GEMM A-stream.
__global__ void build_x_kernel(const float* __restrict__ histN,
                               const float* __restrict__ hist,
                               const float* __restrict__ level,
                               const float* __restrict__ dp,
                               const int* __restrict__ nid,
                               const float* __restrict__ emb,
                               float* __restrict__ x,
                               unsigned short* __restrict__ xb) {
    int bn = blockIdx.x;
    int b = bn / NNODE, n = bn % NNODE;
    float L = level[bn];
    float inv = (L == 0.f) ? 0.f : 1.f / L;
    int id = nid[bn];
    for (int c = threadIdx.x; c < ININ; c += blockDim.x) {
        float v;
        if (c < HH) {
            v = histN[bn * HH + c];
        } else if (c < HH + NNODE * HH) {
            int cj = c - HH; int j = cj / HH; int t = cj - j * HH;
            float a = hist[(b * NNODE + j) * HH + t] * dp[n * NNODE + j];
            float r = (L == 0.f) ? 0.f : (a - L) * inv;
            v = r > 0.f ? r : 0.f;
        } else {
            v = emb[id * DD + (c - (HH + NNODE * HH))];
        }
        x[(size_t)bn * ININ + c]  = v;
        xb[(size_t)bn * ININ + c] = bf16of(v);
    }
}

// forecast_out = (1 + tg_fwd) * fore * level
__global__ void finish_kernel(const float* __restrict__ fore,
                              const float* __restrict__ tgf,
                              const float* __restrict__ level,
                              float* __restrict__ out) {
    int i = blockIdx.x * blockDim.x + threadIdx.x;
    if (i >= BB * NNODE * HORZ) return;
    int bn = i / HORZ;
    out[i] = (1.f + tgf[i]) * fore[i] * level[bn];
}

extern "C" void kernel_launch(void* const* d_in, const int* in_sizes, int n_in,
                              void* d_out, int out_size, void* d_ws, size_t ws_size,
                              hipStream_t stream) {
    (void)in_sizes; (void)n_in; (void)out_size; (void)ws_size;
    const float* history_in = (const float*)d_in[0];
    const int*   node_id    = (const int*)  d_in[1];
    const float* tod        = (const float*)d_in[2];
    const float* emb        = (const float*)d_in[3];
    const float* tg1_w = (const float*)d_in[4];
    const float* tg1_b = (const float*)d_in[5];
    const float* tg2_w = (const float*)d_in[6];
    const float* tg2_b = (const float*)d_in[7];
    const float* tg3_w = (const float*)d_in[8];
    const float* tg3_b = (const float*)d_in[9];
    const float* fc_w0 = (const float*)d_in[10];
    const float* fc_b0 = (const float*)d_in[11];
    const float* fc_wh = (const float*)d_in[12];
    const float* fc_bh = (const float*)d_in[13];
    const float* fore_w = (const float*)d_in[14];
    const float* fore_b = (const float*)d_in[15];
    const float* back_w = (const float*)d_in[16];
    const float* back_b = (const float*)d_in[17];

    const int MM = BB * NNODE;                 // 10400
    float* x    = (float*)d_out;               // backcast region doubles as x
    float* fout = x + (size_t)MM * ININ;       // forecast region

    // Workspace carving (fp32 region, then 16B-aligned bf16 region)
    char* p = (char*)d_ws;
    auto allocF = [&](size_t n) { float* r = (float*)p; p += n * 4; return r; };
    float* dp    = allocF((size_t)NNODE * NNODE);
    float* tgf   = allocF((size_t)MM * HORZ);
    float* tgb   = allocF((size_t)MM * HH);
    float* hist  = allocF((size_t)MM * HH);
    float* level = allocF((size_t)MM);
    float* histN = allocF((size_t)MM * HH);
    float* fore  = allocF((size_t)MM * HORZ);
    auto allocU = [&](size_t n) {
        p = (char*)(((uintptr_t)p + 15) & ~(uintptr_t)15);
        unsigned short* r = (unsigned short*)p; p += n * 2; return r;
    };
    unsigned short* xb    = allocU((size_t)MM * ININ);
    unsigned short* acatb = allocU((size_t)MM * KACAT);
    unsigned short* tgB   = allocU((size_t)MM * HUU);
    unsigned short* h0b   = allocU((size_t)MM * HUU);
    unsigned short* h1b   = allocU((size_t)MM * HUU);
    unsigned short* tg1T  = allocU((size_t)HUU * KACAT);
    unsigned short* tg2T  = allocU((size_t)HORZ * HUU);
    unsigned short* tg3T  = allocU((size_t)HORZ * HUU);
    unsigned short* w0T   = allocU((size_t)NBB * HUU * ININ);
    unsigned short* whT   = allocU((size_t)NBB * 2 * HUU * HUU);
    unsigned short* foreT = allocU((size_t)NBB * HORZ * HUU);
    unsigned short* backT = allocU((size_t)NBB * ININ * HUU);

    dim3 blk(256);
    // Pre-transpose + convert all weights to bf16 [N][K] (tiny, once per call)
    auto wt = [&](const float* W, unsigned short* T, int K, int N, int Kp) {
        int tot = N * Kp;
        wtrans_kernel<<<(tot + 255) / 256, blk, 0, stream>>>(W, T, K, N, Kp);
    };
    wt(tg1_w, tg1T, DD + HH, HUU, KACAT);
    wt(tg2_w, tg2T, HUU, HORZ, HUU);
    wt(tg3_w, tg3T, HUU, HORZ, HUU);
    for (int i = 0; i < NBB; ++i) {
        wt(fc_w0 + (size_t)i * ININ * HUU, w0T + (size_t)i * HUU * ININ, ININ, HUU, ININ);
        wt(fc_wh + (size_t)(2 * i + 0) * HUU * HUU, whT + (size_t)(2 * i + 0) * HUU * HUU, HUU, HUU, HUU);
        wt(fc_wh + (size_t)(2 * i + 1) * HUU * HUU, whT + (size_t)(2 * i + 1) * HUU * HUU, HUU, HUU, HUU);
        wt(fore_w + (size_t)i * HUU * HORZ, foreT + (size_t)i * HORZ * HUU, HUU, HORZ, HUU);
        wt(back_w + (size_t)i * HUU * ININ, backT + (size_t)i * ININ * HUU, HUU, ININ, HUU);
    }

    dp_kernel<<<dim3(NNODE, (NNODE + 255) / 256), blk, 0, stream>>>(emb, dp);
    build_acat_kernel<<<MM, 128, 0, stream>>>(tod, node_id, emb, acatb);

    dim3 g1((MM + 127) / 128, 1);
    // time gate: tg = relu(acat @ tg1_w + b); tg_fwd/tg_bwd = tg @ tg{2,3}_w + b
    gemm_bf16_wmma<<<g1, blk, 0, stream>>>(acatb, tg1T, tg1_b, nullptr, tgB, MM, HUU, KACAT, 0);
    gemm_bf16_wmma<<<g1, blk, 0, stream>>>(tgB, tg2T, tg2_b, tgf, nullptr, MM, HORZ, HUU, 3);
    gemm_bf16_wmma<<<g1, blk, 0, stream>>>(tgB, tg3T, tg3_b, tgb, nullptr, MM, HORZ, HUU, 3);

    gate_hist_kernel<<<(MM + 255) / 256, blk, 0, stream>>>(history_in, tgb, hist, level, histN);
    build_x_kernel<<<MM, blk, 0, stream>>>(histN, hist, level, dp, node_id, emb, x, xb);

    dim3 gb((MM + 127) / 128, (ININ + 127) / 128);
    for (int i = 0; i < NBB; ++i) {
        gemm_bf16_wmma<<<g1, blk, 0, stream>>>(xb, w0T + (size_t)i * HUU * ININ,
                                               fc_b0 + i * HUU, nullptr, h0b, MM, HUU, ININ, 0);
        gemm_bf16_wmma<<<g1, blk, 0, stream>>>(h0b, whT + (size_t)(2 * i + 0) * HUU * HUU,
                                               fc_bh + (2 * i + 0) * HUU, nullptr, h1b, MM, HUU, HUU, 0);
        gemm_bf16_wmma<<<g1, blk, 0, stream>>>(h1b, whT + (size_t)(2 * i + 1) * HUU * HUU,
                                               fc_bh + (2 * i + 1) * HUU, nullptr, h0b, MM, HUU, HUU, 0);
        gemm_bf16_wmma<<<g1, blk, 0, stream>>>(h0b, foreT + (size_t)i * HORZ * HUU,
                                               fore_b + i * HORZ, fore, nullptr, MM, HORZ, HUU,
                                               (i == 0) ? 3 : 1);
        gemm_bf16_wmma<<<gb, blk, 0, stream>>>(h0b, backT + (size_t)i * ININ * HUU,
                                               back_b + i * ININ, x, xb, MM, ININ, HUU, 2);
    }
    finish_kernel<<<(MM * HORZ + 255) / 256, blk, 0, stream>>>(fore, tgf, level, fout);
}
